// TokenizedEmbeddingLSTMRegressor_81458349736264
// MI455X (gfx1250) — compile-verified, matrix operand order checked by reading, not run on hardware
//
#include <hip/hip_runtime.h>
#include <hip/hip_bf16.h>
#include <cstdint>
#include <cstddef>

using bf16 = __bf16;
typedef __attribute__((ext_vector_type(16))) __bf16 v16bf;
typedef __attribute__((ext_vector_type(8)))  float  v8f;
typedef __attribute__((ext_vector_type(8)))  __bf16 bf16x8;

// ---------------- problem dims ----------------
constexpr int Bn   = 32;
constexpr int Tn   = 4096;
constexpr int Cn   = 64;
constexpr int Tp   = 1024;          // T' = T/4
constexpr int Kcb  = 256;           // codebook size
constexpr int Dd   = 64;            // latent dim
constexpr int Hh   = 512;           // LSTM hidden
constexpr int G4H  = 4 * Hh;        // 2048
constexpr int OUTn = 6;
constexpr int Mrows = Bn * Tp;      // 32768

// recurrent kernel config
constexpr int NWG  = 16;            // workgroups in recurrent phase
constexpr int WPAD = Hh + 8;        // padded LDS row stride (elems) -> 1040B, bank-friendly

// ---------------- helpers ----------------
__device__ __forceinline__ v16bf load_frag(const bf16* __restrict__ p0, int strideElems) {
    // 16x32 bf16 tile fragment, row-major source with given element stride.
    int lane = threadIdx.x & 31;
    int r    = lane & 15;
    int kb   = (lane >> 4) << 3;            // 0 or 8
    const bf16* p = p0 + (size_t)r * strideElems + kb;
    bf16x8 lo = *(const bf16x8*)(p);        // k = kb .. kb+7
    bf16x8 hi = *(const bf16x8*)(p + 16);   // k = kb+16 .. kb+23
    v16bf f;
#pragma unroll
    for (int i = 0; i < 8; ++i) { f[i] = lo[i]; f[i + 8] = hi[i]; }
    return f;
}

__device__ __forceinline__ v8f wmma_bf16(v16bf a, v16bf b, v8f c) {
    return __builtin_amdgcn_wmma_f32_16x16x32_bf16(false, a, false, b, (short)0, c, false, false);
}

__device__ __forceinline__ float sigm(float x) { return 1.0f / (1.0f + __expf(-x)); }

// ---- CDNA5 async global->LDS copy (ASYNCcnt path), 16B per lane ----
// GVS addressing: mem = SGPR64 base + VGPR32 offset; dest VGPR = LDS byte address
// (low 32 bits of the generic shared pointer == wave-relative LDS offset).
__device__ __forceinline__ void async_copy16(uint32_t ldsAddr, uint64_t gbase, uint32_t goff) {
    asm volatile("global_load_async_to_lds_b128 %0, %1, %2"
                 :: "v"(ldsAddr), "v"(goff), "s"(gbase)
                 : "memory");
}
__device__ __forceinline__ void async_wait0() {
    asm volatile("s_wait_asynccnt 0x0" ::: "memory");
}

// ---------------- tiny kernels ----------------
__global__ void zero_kernel(unsigned* __restrict__ p, int nWords) {
    int i = blockIdx.x * 256 + threadIdx.x;
    if (i < nWords) p[i] = 0u;
}

__global__ void cvt_bf16_kernel(const float* __restrict__ in, bf16* __restrict__ out, int n) {
    int i = blockIdx.x * 256 + threadIdx.x;
    if (i < n) out[i] = (bf16)in[i];
}

// sig[b,t] = mean_c x[b,t,c]
__global__ void mean_kernel(const float* __restrict__ x, float* __restrict__ sig) {
    int i = blockIdx.x * 256 + threadIdx.x;        // b*Tn + t  (131072)
    const float* p = x + (size_t)i * Cn;
    float s = 0.f;
#pragma unroll
    for (int c = 0; c < Cn; c += 4) {
        float4 v = *(const float4*)(p + c);
        s += v.x + v.y + v.z + v.w;
    }
    sig[i] = s * (1.0f / 64.0f);
}

// conv1: 1->64, k=4, stride 2, SAME (pad_l=1), relu. out [b][c][2048]
__global__ void conv1_kernel(const float* __restrict__ sig, const float* __restrict__ w,
                             const float* __restrict__ bias, float* __restrict__ out) {
    int idx = blockIdx.x * 256 + threadIdx.x;      // 32*64*2048
    int l = idx & 2047;
    int c = (idx >> 11) & 63;
    int b = idx >> 17;
    const float* s = sig + (size_t)b * Tn;
    float acc = bias[c];
    int p0 = 2 * l - 1;
#pragma unroll
    for (int j = 0; j < 4; ++j) {
        int p = p0 + j;
        float v = (p >= 0 && p < Tn) ? s[p] : 0.f;
        acc += w[c * 4 + j] * v;
    }
    out[idx] = fmaxf(acc, 0.f);
}

// conv2: 64->64, k=4, stride 2, SAME (pad_l=1), relu. out [b][c][1024]
__global__ void conv2_kernel(const float* __restrict__ h1, const float* __restrict__ w,
                             const float* __restrict__ bias, float* __restrict__ out) {
    __shared__ float sw[64 * 4];
    int tid = threadIdx.x;
    int lblk = blockIdx.x & 3;
    int c    = (blockIdx.x >> 2) & 63;
    int b    = blockIdx.x >> 8;
    sw[tid] = w[c * 256 + tid];
    __syncthreads();
    int l = lblk * 256 + tid;
    float acc = bias[c];
    int p0 = 2 * l - 1;
    for (int ci = 0; ci < 64; ++ci) {
        const float* row = h1 + ((size_t)b * 64 + ci) * 2048;
#pragma unroll
        for (int j = 0; j < 4; ++j) {
            int p = p0 + j;
            float v = (p >= 0 && p < 2048) ? row[p] : 0.f;
            acc += sw[ci * 4 + j] * v;
        }
    }
    out[((size_t)b * 64 + c) * 1024 + l] = fmaxf(acc, 0.f);
}

// conv3: 64->64, k=3, stride 1, SAME (pad 1). output TRANSPOSED: ze[b][t][d]
__global__ void conv3_kernel(const float* __restrict__ h2, const float* __restrict__ w,
                             const float* __restrict__ bias, float* __restrict__ ze) {
    __shared__ float sw[64 * 3];
    int tid = threadIdx.x;
    int lblk = blockIdx.x & 3;
    int c    = (blockIdx.x >> 2) & 63;
    int b    = blockIdx.x >> 8;
    if (tid < 192) sw[tid] = w[c * 192 + tid];
    __syncthreads();
    int l = lblk * 256 + tid;
    float acc = bias[c];
    for (int ci = 0; ci < 64; ++ci) {
        const float* row = h2 + ((size_t)b * 64 + ci) * 1024;
#pragma unroll
        for (int j = 0; j < 3; ++j) {
            int p = l - 1 + j;
            float v = (p >= 0 && p < 1024) ? row[p] : 0.f;
            acc += sw[ci * 3 + j] * v;
        }
    }
    ze[((size_t)b * 1024 + l) * 64 + c] = acc;
}

// VQ: argmin_k ||z - e_k||^2, output quantized row as bf16
__global__ void vq_kernel(const float* __restrict__ ze, const float* __restrict__ cb,
                          bf16* __restrict__ zq) {
    extern __shared__ char smem_vq[];
    float* scb = (float*)smem_vq;                          // 256 x 64 f32 = 64KB
    int tid = threadIdx.x;

    // async-stage codebook into LDS: 65536B / 256 thr = 16 x 16B each
    {
        uint32_t sbase = (uint32_t)(uintptr_t)scb;
        uint64_t gbase = (uint64_t)(uintptr_t)cb;
        for (int i = tid * 16; i < Kcb * Dd * 4; i += 256 * 16)
            async_copy16(sbase + (uint32_t)i, gbase, (uint32_t)i);
        async_wait0();
    }
    __syncthreads();

    int row = blockIdx.x * 256 + tid;                      // 0..32767
    float z[Dd];
#pragma unroll
    for (int d = 0; d < Dd; d += 4) {
        float4 v = *(const float4*)(ze + (size_t)row * Dd + d);
        z[d] = v.x; z[d + 1] = v.y; z[d + 2] = v.z; z[d + 3] = v.w;
    }
    int best = 0; float bestd = 3.4e38f;
    for (int k = 0; k < Kcb; ++k) {
        const float* e = scb + k * Dd;
        float dot = 0.f, nn = 0.f;
#pragma unroll 8
        for (int d = 0; d < Dd; ++d) { float ev = e[d]; dot += z[d] * ev; nn += ev * ev; }
        float dist = nn - 2.0f * dot;                      // ||z||^2 constant per row
        if (dist < bestd) { bestd = dist; best = k; }
    }
    const float* e = scb + best * Dd;
    for (int d = 0; d < Dd; ++d) zq[(size_t)row * Dd + d] = (bf16)e[d];
}

// ---------------- xg GEMM: Y[M,N] = X[M,K] @ W[N,K]^T + bias, bf16 out ----------------
// K is compile-time (64 or 512): the K-loop fully unrolls, so fragment loads write
// fresh registers (no rotation copies) and the scheduler hoists loads across WMMAs.
template<int K>
__global__ __launch_bounds__(256)
void gemm_xg_kernel(const bf16* __restrict__ X, const bf16* __restrict__ W,
                    const float* __restrict__ bias, bf16* __restrict__ Y) {
    constexpr int N = G4H;
    int wave = threadIdx.x >> 5;
    int lane = threadIdx.x & 31;
    int mBase = blockIdx.x * 16;
    int nBase = blockIdx.y * 512 + wave * 64;

    const bf16* Xp  = X + (size_t)mBase * K;
    const bf16* Wp0 = W + (size_t)(nBase +  0) * K;
    const bf16* Wp1 = W + (size_t)(nBase + 16) * K;
    const bf16* Wp2 = W + (size_t)(nBase + 32) * K;
    const bf16* Wp3 = W + (size_t)(nBase + 48) * K;

    v8f acc[4] = {};
#pragma unroll
    for (int kc = 0; kc < K; kc += 32) {
        v16bf a  = load_frag(Xp  + kc, K);
        v16bf b0 = load_frag(Wp0 + kc, K);
        v16bf b1 = load_frag(Wp1 + kc, K);
        v16bf b2 = load_frag(Wp2 + kc, K);
        v16bf b3 = load_frag(Wp3 + kc, K);
        acc[0] = wmma_bf16(a, b0, acc[0]);
        acc[1] = wmma_bf16(a, b1, acc[1]);
        acc[2] = wmma_bf16(a, b2, acc[2]);
        acc[3] = wmma_bf16(a, b3, acc[3]);
    }

    // epilogue: + bias, store bf16
#pragma unroll
    for (int nt = 0; nt < 4; ++nt) {
        int n  = nBase + nt * 16 + (lane & 15);
        float bn = bias[n];
        int m0 = mBase + (lane >> 4) * 8;
#pragma unroll
        for (int r = 0; r < 8; ++r)
            Y[(size_t)(m0 + r) * N + n] = (bf16)(acc[nt][r] + bn);
    }
}

// ---------------- persistent recurrent LSTM layer ----------------
// 16 WGs x 128 threads. WG wg owns hidden cols [wg*32, wg*32+32), all 4 gates.
// whh slice (128 rows x 512) lives in LDS for all 1024 steps; h double-buffered in global.
__global__ __launch_bounds__(128, 1)
void lstm_rec_kernel(const bf16* __restrict__ xg,     // [B][Tp][2048] (bias included)
                     const float* __restrict__ whh,   // [2048][512] f32
                     bf16* __restrict__ hbuf,         // [2][B][512] double buffer
                     unsigned* __restrict__ cnt,      // grid barrier counter (zeroed)
                     bf16* __restrict__ hout) {       // [B][Tp][512]
    extern __shared__ char smem[];
    bf16*  sW = (bf16*)smem;                               // 128 x WPAD
    bf16*  sH = (bf16*)(smem + (size_t)128 * WPAD * 2);    // 32  x WPAD
    float* sG = (float*)(smem + (size_t)160 * WPAD * 2);   // [4][32][32]
    float* sC = sG + 4 * 32 * 32;                          // [32][32]

    const int wg   = blockIdx.x;        // 0..15
    const int tid  = threadIdx.x;
    const int wave = tid >> 5;          // gate id: 0=i 1=f 2=g 3=o
    const int lane = tid & 31;

    // ---- one-time preload: whh slice -> LDS (bf16), one row per thread ----
    {
        int g  = tid >> 5;
        int jl = tid & 31;
        int grow = g * Hh + wg * 32 + jl;                  // global whh row
        const float* src = whh + (size_t)grow * Hh;
        bf16* dst = sW + (size_t)tid * WPAD;
        for (int k = 0; k < Hh; k += 4) {
            float4 v = *(const float4*)(src + k);
            dst[k]     = (bf16)v.x; dst[k + 1] = (bf16)v.y;
            dst[k + 2] = (bf16)v.z; dst[k + 3] = (bf16)v.w;
        }
    }
    for (int i = tid; i < 32 * 32; i += 128) sC[i] = 0.f;  // c state = 0
    __syncthreads();

    const uint32_t sHbase = (uint32_t)(uintptr_t)sH;

    for (int t = 0; t < Tp; ++t) {
        // ---- async-stage h(t-1) from global (parity (t^1)&1) into LDS ----
        {
            uint64_t gbase = (uint64_t)(uintptr_t)(hbuf + (size_t)((t & 1) ^ 1) * (Bn * Hh));
            for (int i = tid * 8; i < Bn * Hh; i += 128 * 8) {
                int m = i >> 9;               // /512
                int k = i & 511;
                async_copy16(sHbase + (uint32_t)(m * WPAD + k) * 2, gbase, (uint32_t)(i * 2));
            }
            async_wait0();
        }
        __syncthreads();

        // ---- seed accumulators with xg (input gates + bias, precomputed) ----
        v8f acc[2][2];
#pragma unroll
        for (int mt = 0; mt < 2; ++mt)
#pragma unroll
            for (int nt = 0; nt < 2; ++nt) {
                int n = wave * Hh + wg * 32 + nt * 16 + (lane & 15);
                int m0 = mt * 16 + (lane >> 4) * 8;
                v8f a;
#pragma unroll
                for (int r = 0; r < 8; ++r)
                    a[r] = (float)xg[((size_t)(m0 + r) * Tp + t) * G4H + n];
                acc[mt][nt] = a;
            }

        // ---- gates += h(t-1) @ whh_slice^T (WMMA, all operands in LDS) ----
#pragma unroll 4
        for (int kc = 0; kc < Hh / 32; ++kc) {
            v16bf a0 = load_frag(sH + kc * 32, WPAD);
            v16bf a1 = load_frag(sH + (size_t)16 * WPAD + kc * 32, WPAD);
            v16bf b0 = load_frag(sW + (size_t)(wave * 32) * WPAD + kc * 32, WPAD);
            v16bf b1 = load_frag(sW + (size_t)(wave * 32 + 16) * WPAD + kc * 32, WPAD);
            acc[0][0] = wmma_bf16(a0, b0, acc[0][0]);
            acc[0][1] = wmma_bf16(a0, b1, acc[0][1]);
            acc[1][0] = wmma_bf16(a1, b0, acc[1][0]);
            acc[1][1] = wmma_bf16(a1, b1, acc[1][1]);
        }

        // ---- exchange gate pre-activations through LDS ----
#pragma unroll
        for (int mt = 0; mt < 2; ++mt)
#pragma unroll
            for (int nt = 0; nt < 2; ++nt) {
                int nl = nt * 16 + (lane & 15);
                int m0 = mt * 16 + (lane >> 4) * 8;
#pragma unroll
                for (int r = 0; r < 8; ++r)
                    sG[((wave << 5) + m0 + r) * 32 + nl] = acc[mt][nt][r];
            }
        __syncthreads();

        // ---- elementwise cell update: 1024 cells / 128 threads ----
        {
            int m  = tid >> 2;
            int j0 = (tid & 3) * 8;
            bf16* hdst = hbuf + (size_t)(t & 1) * (Bn * Hh);
#pragma unroll
            for (int e = 0; e < 8; ++e) {
                int jl = j0 + e;
                float iv = sG[(0 * 32 + m) * 32 + jl];
                float fv = sG[(1 * 32 + m) * 32 + jl];
                float gv = sG[(2 * 32 + m) * 32 + jl];
                float ov = sG[(3 * 32 + m) * 32 + jl];
                float c  = sigm(fv) * sC[m * 32 + jl] + sigm(iv) * tanhf(gv);
                sC[m * 32 + jl] = c;
                float h = sigm(ov) * tanhf(c);
                int jg = wg * 32 + jl;
                hdst[m * Hh + jg] = (bf16)h;
                hout[((size_t)m * Tp + t) * Hh + jg] = (bf16)h;
            }
        }

        // ---- device-wide step barrier (16 WGs) ----
        __threadfence();
        __syncthreads();
        if (t < Tp - 1) {
            if (tid == 0) {
                unsigned goal = (unsigned)NWG * (unsigned)(t + 1);
                atomicAdd(cnt, 1u);
                while (atomicAdd(cnt, 0u) < goal) __builtin_amdgcn_s_sleep(1);
            }
            __syncthreads();
            __threadfence();
        }
    }
}

// ---------------- output head ----------------
__global__ void head_kernel(const bf16* __restrict__ h, const float* __restrict__ wout,
                            const float* __restrict__ bout, float* __restrict__ out) {
    int idx = blockIdx.x * 256 + threadIdx.x;       // (b*Tp+t)*6 + o, 196608 total
    if (idx >= Mrows * OUTn) return;
    int o  = idx % OUTn;
    int rt = idx / OUTn;
    const bf16*  hp = h + (size_t)rt * Hh;
    const float* wp = wout + o * Hh;
    float acc = bout[o];
    for (int k = 0; k < Hh; ++k) acc += (float)hp[k] * wp[k];
    out[idx] = acc;
}

// ---------------- host orchestration ----------------
static constexpr size_t SIG_B   = (size_t)Bn * Tn * 4;            // 512 KB
static constexpr size_t H1_B    = (size_t)Bn * 64 * 2048 * 4;     // 16 MB
static constexpr size_t H2_B    = (size_t)Bn * 64 * 1024 * 4;     // 8 MB
static constexpr size_t ZE_B    = (size_t)Mrows * Dd * 4;         // 8 MB
static constexpr size_t ZQ_B    = (size_t)Mrows * Dd * 2;         // 4 MB
static constexpr size_t WIH0_B  = (size_t)G4H * Dd * 2;
static constexpr size_t WIH1_B  = (size_t)G4H * Hh * 2;
static constexpr size_t XG_B    = (size_t)Mrows * G4H * 2;        // 128 MB (reused by both layers)
static constexpr size_t HOUT_B  = (size_t)Mrows * Hh * 2;         // 32 MB
static constexpr size_t HBUF_B  = (size_t)2 * Bn * Hh * 2;        // 64 KB

static constexpr size_t OFF_SIG   = 0;
static constexpr size_t OFF_H1    = OFF_SIG   + SIG_B;
static constexpr size_t OFF_H2    = OFF_H1    + H1_B;
static constexpr size_t OFF_ZE    = OFF_H2    + H2_B;
static constexpr size_t OFF_ZQ    = OFF_ZE    + ZE_B;
static constexpr size_t OFF_WIH0  = OFF_ZQ    + ZQ_B;
static constexpr size_t OFF_WIH1  = OFF_WIH0  + WIH0_B;
static constexpr size_t OFF_XG    = OFF_WIH1  + WIH1_B;
static constexpr size_t OFF_H0OUT = OFF_XG    + XG_B;
static constexpr size_t OFF_H1OUT = OFF_H0OUT + HOUT_B;
static constexpr size_t OFF_HBUF0 = OFF_H1OUT + HOUT_B;
static constexpr size_t OFF_HBUF1 = OFF_HBUF0 + HBUF_B;
static constexpr size_t OFF_CNT   = OFF_HBUF1 + HBUF_B;           // 256 B (2 counters)

static constexpr int LSTM_SMEM = 160 * WPAD * 2 + 4 * 32 * 32 * 4 + 32 * 32 * 4; // 186,880 B
static constexpr int VQ_SMEM   = Kcb * Dd * 4;                                   // 65,536 B

extern "C" void kernel_launch(void* const* d_in, const int* in_sizes, int n_in,
                              void* d_out, int out_size, void* d_ws, size_t ws_size,
                              hipStream_t stream) {
    (void)in_sizes; (void)n_in; (void)out_size; (void)ws_size;

    const float* x    = (const float*)d_in[0];
    const float* cw1  = (const float*)d_in[1];
    const float* cb1  = (const float*)d_in[2];
    const float* cw2  = (const float*)d_in[3];
    const float* cb2  = (const float*)d_in[4];
    const float* cw3  = (const float*)d_in[5];
    const float* cb3  = (const float*)d_in[6];
    const float* cbk  = (const float*)d_in[7];
    const float* wih0 = (const float*)d_in[8];
    const float* whh0 = (const float*)d_in[9];
    const float* bl0  = (const float*)d_in[10];
    const float* wih1 = (const float*)d_in[11];
    const float* whh1 = (const float*)d_in[12];
    const float* bl1  = (const float*)d_in[13];
    const float* wout = (const float*)d_in[14];
    const float* bout = (const float*)d_in[15];

    uint8_t* ws = (uint8_t*)d_ws;
    float* sig    = (float*)(ws + OFF_SIG);
    float* h1     = (float*)(ws + OFF_H1);
    float* h2     = (float*)(ws + OFF_H2);
    float* ze     = (float*)(ws + OFF_ZE);
    bf16*  zq     = (bf16*) (ws + OFF_ZQ);
    bf16*  wih0b  = (bf16*) (ws + OFF_WIH0);
    bf16*  wih1b  = (bf16*) (ws + OFF_WIH1);
    bf16*  xg     = (bf16*) (ws + OFF_XG);
    bf16*  h0out  = (bf16*) (ws + OFF_H0OUT);
    bf16*  h1out  = (bf16*) (ws + OFF_H1OUT);
    bf16*  hbuf0  = (bf16*) (ws + OFF_HBUF0);
    bf16*  hbuf1  = (bf16*) (ws + OFF_HBUF1);
    unsigned* cnt0 = (unsigned*)(ws + OFF_CNT);
    unsigned* cnt1 = (unsigned*)(ws + OFF_CNT + 128);

    // allow big dynamic LDS (320KB/WGP on gfx1250); harmless if already set
    (void)hipFuncSetAttribute((const void*)lstm_rec_kernel,
                              hipFuncAttributeMaxDynamicSharedMemorySize, LSTM_SMEM);
    (void)hipFuncSetAttribute((const void*)vq_kernel,
                              hipFuncAttributeMaxDynamicSharedMemorySize, VQ_SMEM);

    // zero h double-buffers + barrier counters (contiguous region)
    {
        int nWords = (int)((HBUF_B * 2 + 256) / 4);
        zero_kernel<<<(nWords + 255) / 256, 256, 0, stream>>>((unsigned*)(ws + OFF_HBUF0), nWords);
    }

    // tokenizer
    mean_kernel <<<(Bn * Tn) / 256, 256, 0, stream>>>(x, sig);
    conv1_kernel<<<(Bn * 64 * 2048) / 256, 256, 0, stream>>>(sig, cw1, cb1, h1);
    conv2_kernel<<<Bn * 64 * 4, 256, 0, stream>>>(h1, cw2, cb2, h2);
    conv3_kernel<<<Bn * 64 * 4, 256, 0, stream>>>(h2, cw3, cb3, ze);
    vq_kernel   <<<Mrows / 256, 256, VQ_SMEM, stream>>>(ze, cbk, zq);

    // weight conversions
    cvt_bf16_kernel<<<(G4H * Dd + 255) / 256, 256, 0, stream>>>(wih0, wih0b, G4H * Dd);
    cvt_bf16_kernel<<<(G4H * Hh + 255) / 256, 256, 0, stream>>>(wih1, wih1b, G4H * Hh);

    // layer 0: xg = z @ wih0^T + b ; then recurrence
    gemm_xg_kernel<Dd><<<dim3(Mrows / 16, G4H / 512), 256, 0, stream>>>(zq, wih0b, bl0, xg);
    lstm_rec_kernel<<<NWG, 128, LSTM_SMEM, stream>>>(xg, whh0, hbuf0, cnt0, h0out);

    // layer 1: xg = h0 @ wih1^T + b ; then recurrence
    gemm_xg_kernel<Hh><<<dim3(Mrows / 16, G4H / 512), 256, 0, stream>>>(h0out, wih1b, bl1, xg);
    lstm_rec_kernel<<<NWG, 128, LSTM_SMEM, stream>>>(xg, whh1, hbuf1, cnt1, h1out);

    // head
    head_kernel<<<(Mrows * OUTn + 255) / 256, 256, 0, stream>>>(h1out, wout, bout, (float*)d_out);
}
